// SDUNET_24661702214110
// MI455X (gfx1250) — matrix operand order repaired and checked by reading, not compiled, not run on hardware
//
#include <hip/hip_runtime.h>
#include <hip/hip_bf16.h>
#include <math.h>

typedef _Float16 h16;
typedef __attribute__((ext_vector_type(16))) _Float16 v16h;
typedef __attribute__((ext_vector_type(8)))  float    v8f;

#define WMMA_F16(a, b, c) \
    __builtin_amdgcn_wmma_f32_16x16x32_f16(false, (a), false, (b), (short)0, (c), false, false)

// ---------------------------------------------------------------------------
// Fragment loaders (wave32; contiguous k order, same in A and B), vectorized as 16B LDS loads.
// Requirements: base 16B-aligned, ld % 8 == 0  (we use ld = 40 / 64 / 96).
//
// A 16x32 f16: lane L (and L+16) holds row M=L&15.
//   halves 0..15 = row[(L<16?0:16) .. +15]   (contiguous)
// ---------------------------------------------------------------------------
__device__ __forceinline__ v16h load_a_frag(const h16* base, int ld, int lane) {
    int row = lane & 15;
    int hi16 = ((lane >> 4) & 1) * 16;
    const h16* rp = base + row * ld;
    union { v16h v; uint4 q[2]; } u;
    u.q[0] = *(const uint4*)(rp + hi16);
    u.q[1] = *(const uint4*)(rp + hi16 + 8);
    return u.v;
}

// B 32x16 f16 from an n-major (transposed) tile Bt[n][k]:
// lane L holds column n=L&15, K = (L<16?0:16) + j, j=0..15 contiguous.
__device__ __forceinline__ v16h load_bt_frag(const h16* base, int ld, int lane) {
    int col = lane & 15;
    int kb  = (lane >> 4) ? 16 : 0;
    const h16* rp = base + col * ld + kb;
    union { v16h v; uint4 q[2]; } u;
    u.q[0] = *(const uint4*)(rp);
    u.q[1] = *(const uint4*)(rp + 8);
    return u.v;
}

// ---------------------------------------------------------------------------
// GroupNorm (32 groups of 10 channels over 4096 pixels)
// ---------------------------------------------------------------------------
__global__ void sd_gn_stats_kernel(const float* __restrict__ x, float* __restrict__ stats) {
    __shared__ float ssum[256], ssq[256];
    int g = blockIdx.x, t = threadIdx.x;
    const int NEL = 10 * 4096;
    const float* base = x + (size_t)g * NEL;
    float s = 0.f, q = 0.f;
    for (int i = t; i < NEL; i += 256) { float v = base[i]; s += v; q += v * v; }
    ssum[t] = s; ssq[t] = q; __syncthreads();
    for (int st = 128; st > 0; st >>= 1) {
        if (t < st) { ssum[t] += ssum[t + st]; ssq[t] += ssq[t + st]; }
        __syncthreads();
    }
    if (t == 0) {
        float mu  = ssum[0] / NEL;
        float var = ssq[0] / NEL - mu * mu;
        stats[2 * g] = mu; stats[2 * g + 1] = rsqrtf(var + 1e-6f);
    }
}

// normalize + transpose CHW -> (HW, C) f16
__global__ void sd_gn_apply_kernel(const float* __restrict__ x, const float* __restrict__ stats,
                                   const float* __restrict__ gam, const float* __restrict__ bet,
                                   h16* __restrict__ G) {
    int e = blockIdx.x * 256 + threadIdx.x;
    if (e >= 320 * 4096) return;
    int c = e >> 12, hw = e & 4095;
    int g = c / 10;
    float v = (x[e] - stats[2 * g]) * stats[2 * g + 1] * gam[c] + bet[c];
    G[(size_t)hw * 320 + c] = (h16)v;
}

// plain f32 -> f16 (for activations used as A operand)
__global__ void sd_f32_to_f16_kernel(const float* __restrict__ s, h16* __restrict__ d, int n) {
    int i = blockIdx.x * 256 + threadIdx.x;
    if (i < n) d[i] = (h16)s[i];
}

// f32 [K,N] -> f16 transposed [N,K] (GEMM-ready B operand; coalesced writes)
__global__ void sd_cvt_t_kernel(const float* __restrict__ w, h16* __restrict__ wT, int K, int N) {
    int e = blockIdx.x * 256 + threadIdx.x;
    if (e >= K * N) return;
    int n = e / K, k = e - n * K;
    wT[e] = (h16)w[(size_t)k * N + n];
}

// ---------------------------------------------------------------------------
// LayerNorm over D=320, one wave32 per token
// ---------------------------------------------------------------------------
__global__ void sd_ln_kernel(const float* __restrict__ x, const float* __restrict__ g,
                             const float* __restrict__ b, h16* __restrict__ out) {
    int row = blockIdx.x, lane = threadIdx.x;
    const float* xr = x + (size_t)row * 320;
    float vals[10], s = 0.f, q = 0.f;
#pragma unroll
    for (int i = 0; i < 10; ++i) { float v = xr[lane + 32 * i]; vals[i] = v; s += v; q += v * v; }
#pragma unroll
    for (int o = 1; o < 32; o <<= 1) { s += __shfl_xor(s, o, 32); q += __shfl_xor(q, o, 32); }
    float mu = s * (1.f / 320.f);
    float r  = rsqrtf(q * (1.f / 320.f) - mu * mu + 1e-5f);
    h16* orow = out + (size_t)row * 320;
#pragma unroll
    for (int i = 0; i < 10; ++i) {
        int c = lane + 32 * i;
        orow[c] = (h16)((vals[i] - mu) * r * g[c] + b[c]);
    }
}

// ---------------------------------------------------------------------------
// Generic WMMA GEMM: C[M,N] = A[M,K]h16 @ BT[N,K]h16^T (+bias)(+resid)
// BT is pre-transposed (n-major), so both staging copies are contiguous b128.
// 128x64 workgroup tile, 8 waves, each wave 32x32 (4 WMMAs per K-step).
// Requirements: K % 32 == 0, N % 64 == 0 (true at every call site).
// A rows beyond M are address-clamped; their results are discarded in the
// epilogue, so no zero-fill or branch is needed.
// ---------------------------------------------------------------------------
__global__ __launch_bounds__(256) void sd_gemm_kernel(
    const h16* __restrict__ A, const h16* __restrict__ BT,
    const float* __restrict__ bias, const float* __restrict__ resid,
    float* __restrict__ outF, h16* __restrict__ outH, int M, int N, int K) {
    __shared__ __align__(16) h16 As[128 * 40];   // row-major, ld=40
    __shared__ __align__(16) h16 Bts[64 * 40];   // n-major, ld=40
    int tid = threadIdx.x, lane = tid & 31, wid = tid >> 5;
    int m0 = blockIdx.y * 128, n0 = blockIdx.x * 64;
    int msub = (wid & 3) * 32, nsub = (wid >> 2) * 32;
    v8f c00 = {}, c01 = {}, c10 = {}, c11 = {};
    for (int k0 = 0; k0 < K; k0 += 32) {
        __syncthreads();
        // --- stage A: 128x32 halfs = 512 uint4, 2 per thread (rows clamped) ---
#pragma unroll
        for (int u2 = 0; u2 < 2; ++u2) {
            int u = tid + 256 * u2;
            int r = u >> 2, g = u & 3;
            int gm = m0 + r; if (gm > M - 1) gm = M - 1;
            *(uint4*)&As[r * 40 + g * 8] = *(const uint4*)&A[(size_t)gm * K + k0 + g * 8];
        }
        // --- stage B: 64n x 32k halfs = 256 uint4, 1 per thread ---
        {
            int n = tid >> 2, g = tid & 3;
            *(uint4*)&Bts[n * 40 + g * 8] = *(const uint4*)&BT[(size_t)(n0 + n) * K + k0 + g * 8];
        }
        if (k0 + 32 < K) {
            int pm = m0 + (tid >> 1); if (pm > M - 1) pm = M - 1;
            __builtin_prefetch(&A[(size_t)pm * K + k0 + 32], 0, 1);
            __builtin_prefetch(&BT[(size_t)(n0 + (tid >> 2)) * K + k0 + 32], 0, 1);
        }
        __syncthreads();
        v16h aF0 = load_a_frag(&As[msub * 40], 40, lane);
        v16h aF1 = load_a_frag(&As[(msub + 16) * 40], 40, lane);
        v16h b0  = load_bt_frag(&Bts[nsub * 40], 40, lane);
        v16h b1  = load_bt_frag(&Bts[(nsub + 16) * 40], 40, lane);
        c00 = WMMA_F16(aF0, b0, c00);
        c01 = WMMA_F16(aF0, b1, c01);
        c10 = WMMA_F16(aF1, b0, c10);
        c11 = WMMA_F16(aF1, b1, c11);
    }
    int hi = lane >> 4, col = lane & 15;
#pragma unroll
    for (int mt = 0; mt < 2; ++mt) {
#pragma unroll
        for (int r = 0; r < 8; ++r) {
            int gm = m0 + msub + mt * 16 + r + 8 * hi;
            if (gm >= M) continue;
#pragma unroll
            for (int ct = 0; ct < 2; ++ct) {
                int gn = n0 + nsub + 16 * ct + col;
                float v;
                if (mt == 0) v = (ct == 0) ? c00[r] : c01[r];
                else         v = (ct == 0) ? c10[r] : c11[r];
                if (bias)  v += bias[gn];
                size_t o = (size_t)gm * N + gn;
                if (resid) v += resid[o];
                if (outF)  outF[o] = v;
                if (outH)  outH[o] = (h16)v;
            }
        }
    }
}

// ---------------------------------------------------------------------------
// Repack [S,320] -> per-head padded layouts (dh 40 -> 64, zeros in pad)
// Qp/Kp: [head][S][64] (natural; K is the *n*-operand of QK^T so n-major
// already). Vt: [head][64][S] (transposed; V's n-dim is dh for PV).
// ---------------------------------------------------------------------------
__global__ void sd_repack_qkv1_kernel(const h16* __restrict__ q, const h16* __restrict__ k,
                                      const h16* __restrict__ v, h16* __restrict__ Qp,
                                      h16* __restrict__ Kp, h16* __restrict__ Vt) {
    int e = blockIdx.x * 256 + threadIdx.x;
    if (e >= 8 * 4096 * 64) return;
    int head = e >> 18, rem = e & 262143;
    int s = rem >> 6, d = rem & 63;
    h16 qv = (h16)0.f, kv = (h16)0.f, vv = (h16)0.f;
    if (d < 40) {
        size_t src = (size_t)s * 320 + head * 40 + d;
        qv = q[src]; kv = k[src]; vv = v[src];
    }
    Qp[e] = qv; Kp[e] = kv;
    Vt[((size_t)head << 18) + ((size_t)d << 12) + s] = vv;
}

__global__ void sd_repack_q_kernel(const h16* __restrict__ q, h16* __restrict__ Qp) {
    int e = blockIdx.x * 256 + threadIdx.x;
    if (e >= 8 * 4096 * 64) return;
    int head = e >> 18, rem = e & 262143;
    int s = rem >> 6, d = rem & 63;
    h16 qv = (h16)0.f;
    if (d < 40) qv = q[(size_t)s * 320 + head * 40 + d];
    Qp[e] = qv;
}

// cross-attn: 77 keys padded to 96. K2p: [head][96][64]; V2t: [head][64][96]
__global__ void sd_repack_kv2_kernel(const h16* __restrict__ k, const h16* __restrict__ v,
                                     h16* __restrict__ K2p, h16* __restrict__ V2t) {
    int e = blockIdx.x * 256 + threadIdx.x;
    if (e >= 8 * 96 * 64) return;
    int head = e / (96 * 64), rem = e % (96 * 64);
    int kk = rem >> 6, d = rem & 63;
    h16 kv = (h16)0.f, vv = (h16)0.f;
    if (kk < 77 && d < 40) {
        size_t src = (size_t)kk * 320 + head * 40 + d;
        kv = k[src]; vv = v[src];
    }
    K2p[e] = kv;
    V2t[(size_t)head * (64 * 96) + d * 96 + kk] = vv;
}

// ---------------------------------------------------------------------------
// Flash attention, self-attn: S=4096 keys streamed in 64-key chunks.
// Grid (64 q-blocks, 8 heads), 128 threads = 4 waves, each wave 16 q-rows.
// ---------------------------------------------------------------------------
__global__ __launch_bounds__(128) void sd_attn1_kernel(const h16* __restrict__ Qp,
                                                       const h16* __restrict__ Kp,
                                                       const h16* __restrict__ Vt,
                                                       h16* __restrict__ O) {
    __shared__ __align__(16) h16 Qs[64 * 64];    // [qrow][d]
    __shared__ __align__(16) h16 Kls[64 * 64];   // [key][d]   (n-major for QK^T)
    __shared__ __align__(16) h16 Vts[64 * 64];   // [d][key]   (n-major for PV)
    __shared__ __align__(16) h16 Ps[64 * 64];    // per-wave 16x64 P tiles
    const float SC = 0.15811388300841898f;       // 40^-0.5
    int tid = threadIdx.x, lane = tid & 31, wid = tid >> 5;
    int head = blockIdx.y, qbase = blockIdx.x * 64;
    const h16* Qg  = Qp + ((size_t)head << 18) + ((size_t)qbase << 6);
    const h16* Kh  = Kp + ((size_t)head << 18);
    const h16* Vth = Vt + ((size_t)head << 18);
    for (int u = tid; u < 512; u += 128) {
        int r = u >> 3, g = u & 7;
        *(uint4*)&Qs[r * 64 + g * 8] = *(const uint4*)&Qg[(size_t)r * 64 + g * 8];
    }
    float m_i[8], l_i[8];
#pragma unroll
    for (int r = 0; r < 8; ++r) { m_i[r] = -1e30f; l_i[r] = 0.f; }
    v8f o0 = {}, o1 = {}, o2 = {}, o3 = {};
    int hi = lane >> 4, col = lane & 15;
    h16* Psw = Ps + wid * 16 * 64;
    for (int j0 = 0; j0 < 4096; j0 += 64) {
        __syncthreads();
        for (int u = tid; u < 512; u += 128) {
            int r = u >> 3, g = u & 7;
            *(uint4*)&Kls[r * 64 + g * 8] = *(const uint4*)&Kh[(size_t)(j0 + r) * 64 + g * 8];
            *(uint4*)&Vts[r * 64 + g * 8] = *(const uint4*)&Vth[(size_t)r * 4096 + j0 + g * 8];
        }
        __syncthreads();
        v8f s0 = {}, s1 = {}, s2 = {}, s3 = {};
#pragma unroll
        for (int ks = 0; ks < 64; ks += 32) {
            v16h aF = load_a_frag(&Qs[(wid * 16) * 64 + ks], 64, lane);
            s0 = WMMA_F16(aF, load_bt_frag(&Kls[0 * 16 * 64 + ks], 64, lane), s0);
            s1 = WMMA_F16(aF, load_bt_frag(&Kls[1 * 16 * 64 + ks], 64, lane), s1);
            s2 = WMMA_F16(aF, load_bt_frag(&Kls[2 * 16 * 64 + ks], 64, lane), s2);
            s3 = WMMA_F16(aF, load_bt_frag(&Kls[3 * 16 * 64 + ks], 64, lane), s3);
        }
        float mnew[8], corr[8];
#pragma unroll
        for (int r = 0; r < 8; ++r) {
            s0[r] *= SC; s1[r] *= SC; s2[r] *= SC; s3[r] *= SC;
            float mt = fmaxf(fmaxf(s0[r], s1[r]), fmaxf(s2[r], s3[r]));
#pragma unroll
            for (int o = 1; o < 16; o <<= 1) mt = fmaxf(mt, __shfl_xor(mt, o, 32));
            mnew[r] = fmaxf(m_i[r], mt);
            corr[r] = __expf(m_i[r] - mnew[r]);
            m_i[r]  = mnew[r];
        }
#pragma unroll
        for (int r = 0; r < 8; ++r) {
            int row = r + 8 * hi;
            float p0 = __expf(s0[r] - mnew[r]);
            float p1 = __expf(s1[r] - mnew[r]);
            float p2 = __expf(s2[r] - mnew[r]);
            float p3 = __expf(s3[r] - mnew[r]);
            float ls = p0 + p1 + p2 + p3;
#pragma unroll
            for (int o = 1; o < 16; o <<= 1) ls += __shfl_xor(ls, o, 32);
            l_i[r] = l_i[r] * corr[r] + ls;
            Psw[row * 64 + 0  + col] = (h16)p0;
            Psw[row * 64 + 16 + col] = (h16)p1;
            Psw[row * 64 + 32 + col] = (h16)p2;
            Psw[row * 64 + 48 + col] = (h16)p3;
            o0[r] *= corr[r]; o1[r] *= corr[r]; o2[r] *= corr[r]; o3[r] *= corr[r];
        }
        __syncthreads();
#pragma unroll
        for (int ks = 0; ks < 64; ks += 32) {
            v16h aF = load_a_frag(&Psw[ks], 64, lane);
            o0 = WMMA_F16(aF, load_bt_frag(&Vts[0 * 16 * 64 + ks], 64, lane), o0);
            o1 = WMMA_F16(aF, load_bt_frag(&Vts[1 * 16 * 64 + ks], 64, lane), o1);
            o2 = WMMA_F16(aF, load_bt_frag(&Vts[2 * 16 * 64 + ks], 64, lane), o2);
            o3 = WMMA_F16(aF, load_bt_frag(&Vts[3 * 16 * 64 + ks], 64, lane), o3);
        }
    }
#pragma unroll
    for (int r = 0; r < 8; ++r) {
        float inv = 1.f / l_i[r];
        int row = r + 8 * hi;
        int tok = qbase + wid * 16 + row;
        h16* orow = O + (size_t)tok * 320 + head * 40;
        orow[col]      = (h16)(o0[r] * inv);
        orow[16 + col] = (h16)(o1[r] * inv);
        if (32 + col < 40) orow[32 + col] = (h16)(o2[r] * inv);
        // d = 48..63 is zero padding; never stored
    }
}

// ---------------------------------------------------------------------------
// Cross-attention: 77 keys (padded to 96), single pass masked softmax.
// ---------------------------------------------------------------------------
__global__ __launch_bounds__(128) void sd_attn2_kernel(const h16* __restrict__ Qp,
                                                       const h16* __restrict__ K2p,
                                                       const h16* __restrict__ V2t,
                                                       h16* __restrict__ O) {
    __shared__ __align__(16) h16 Qs[64 * 64];    // [qrow][d]
    __shared__ __align__(16) h16 Ks2[96 * 64];   // [key][d]
    __shared__ __align__(16) h16 Vt2[64 * 96];   // [d][key]
    __shared__ __align__(16) h16 Ps[64 * 96];
    const float SC = 0.15811388300841898f;
    int tid = threadIdx.x, lane = tid & 31, wid = tid >> 5;
    int head = blockIdx.y, qbase = blockIdx.x * 64;
    const h16* Qg = Qp + ((size_t)head << 18) + ((size_t)qbase << 6);
    for (int u = tid; u < 512; u += 128) {
        int r = u >> 3, g = u & 7;
        *(uint4*)&Qs[r * 64 + g * 8] = *(const uint4*)&Qg[(size_t)r * 64 + g * 8];
    }
    for (int u = tid; u < 768; u += 128) {
        *(uint4*)&Ks2[u * 8] = *(const uint4*)&K2p[(size_t)head * 6144 + u * 8];
        *(uint4*)&Vt2[u * 8] = *(const uint4*)&V2t[(size_t)head * 6144 + u * 8];
    }
    __syncthreads();
    v8f sa[6];
#pragma unroll
    for (int nt = 0; nt < 6; ++nt) sa[nt] = (v8f){};
#pragma unroll
    for (int ks = 0; ks < 64; ks += 32) {
        v16h aF = load_a_frag(&Qs[(wid * 16) * 64 + ks], 64, lane);
#pragma unroll
        for (int nt = 0; nt < 6; ++nt)
            sa[nt] = WMMA_F16(aF, load_bt_frag(&Ks2[(nt * 16) * 64 + ks], 64, lane), sa[nt]);
    }
    int hi = lane >> 4, col = lane & 15;
    h16* Psw = Ps + wid * 16 * 96;
    float l_i[8];
#pragma unroll
    for (int r = 0; r < 8; ++r) {
        float mt = -1e30f;
#pragma unroll
        for (int nt = 0; nt < 6; ++nt) {
            float v = sa[nt][r] * SC;
            if (nt * 16 + col >= 77) v = -1e30f;
            sa[nt][r] = v;
            mt = fmaxf(mt, v);
        }
#pragma unroll
        for (int o = 1; o < 16; o <<= 1) mt = fmaxf(mt, __shfl_xor(mt, o, 32));
        int row = r + 8 * hi;
        float ls = 0.f;
#pragma unroll
        for (int nt = 0; nt < 6; ++nt) {
            float p = __expf(sa[nt][r] - mt);
            ls += p;
            Psw[row * 96 + nt * 16 + col] = (h16)p;
        }
#pragma unroll
        for (int o = 1; o < 16; o <<= 1) ls += __shfl_xor(ls, o, 32);
        l_i[r] = ls;
    }
    __syncthreads();
    v8f o0 = {}, o1 = {}, o2 = {}, o3 = {};
#pragma unroll
    for (int ks = 0; ks < 96; ks += 32) {
        v16h aF = load_a_frag(&Psw[ks], 96, lane);
        o0 = WMMA_F16(aF, load_bt_frag(&Vt2[(0 * 16) * 96 + ks], 96, lane), o0);
        o1 = WMMA_F16(aF, load_bt_frag(&Vt2[(1 * 16) * 96 + ks], 96, lane), o1);
        o2 = WMMA_F16(aF, load_bt_frag(&Vt2[(2 * 16) * 96 + ks], 96, lane), o2);
        o3 = WMMA_F16(aF, load_bt_frag(&Vt2[(3 * 16) * 96 + ks], 96, lane), o3);
    }
#pragma unroll
    for (int r = 0; r < 8; ++r) {
        float inv = 1.f / l_i[r];
        int row = r + 8 * hi;
        int tok = qbase + wid * 16 + row;
        h16* orow = O + (size_t)tok * 320 + head * 40;
        orow[col]      = (h16)(o0[r] * inv);
        orow[16 + col] = (h16)(o1[r] * inv);
        if (32 + col < 40) orow[32 + col] = (h16)(o2[r] * inv);
    }
}

// ---------------------------------------------------------------------------
// GEGLU: a * gelu(gate), exact erf gelu
// ---------------------------------------------------------------------------
__global__ void sd_geglu_kernel(const float* __restrict__ hg, h16* __restrict__ fg) {
    int e = blockIdx.x * 256 + threadIdx.x;
    if (e >= 4096 * 1280) return;
    int s = e / 1280, i = e % 1280;
    float a = hg[(size_t)s * 2560 + i];
    float g = hg[(size_t)s * 2560 + 1280 + i];
    float ge = 0.5f * g * (1.f + erff(g * 0.70710678118654752f));
    fg[e] = (h16)(a * ge);
}

// out[c,hw] = P[hw,c] + residual[c,hw]
__global__ void sd_final_kernel(const float* __restrict__ P, const float* __restrict__ X,
                                float* __restrict__ out) {
    int e = blockIdx.x * 256 + threadIdx.x;
    if (e >= 320 * 4096) return;
    int c = e >> 12, hw = e & 4095;
    out[e] = P[(size_t)hw * 320 + c] + X[e];
}

// ===========================================================================
extern "C" void kernel_launch(void* const* d_in, const int* in_sizes, int n_in,
                              void* d_out, int out_size, void* d_ws, size_t ws_size,
                              hipStream_t stream) {
    (void)in_sizes; (void)n_in; (void)out_size; (void)ws_size;
    const float* X    = (const float*)d_in[0];   // [320,64,64]
    const float* CTX  = (const float*)d_in[1];   // [77,768]
    const float* gn_g = (const float*)d_in[2];
    const float* gn_b = (const float*)d_in[3];
    const float* pi_w = (const float*)d_in[4];
    const float* pi_b = (const float*)d_in[5];
    const float* n1_g = (const float*)d_in[6];
    const float* n1_b = (const float*)d_in[7];
    const float* a1_q = (const float*)d_in[8];
    const float* a1_k = (const float*)d_in[9];
    const float* a1_v = (const float*)d_in[10];
    const float* a1_o = (const float*)d_in[11];
    const float* a1_ob= (const float*)d_in[12];
    const float* n2_g = (const float*)d_in[13];
    const float* n2_b = (const float*)d_in[14];
    const float* a2_q = (const float*)d_in[15];
    const float* a2_k = (const float*)d_in[16];
    const float* a2_v = (const float*)d_in[17];
    const float* a2_o = (const float*)d_in[18];
    const float* a2_ob= (const float*)d_in[19];
    const float* n3_g = (const float*)d_in[20];
    const float* n3_b = (const float*)d_in[21];
    const float* ffw1 = (const float*)d_in[22];
    const float* ffb1 = (const float*)d_in[23];
    const float* ffw2 = (const float*)d_in[24];
    const float* ffb2 = (const float*)d_in[25];
    const float* po_w = (const float*)d_in[26];
    const float* po_b = (const float*)d_in[27];
    float* OUT = (float*)d_out;

    char* ws = (char*)d_ws;
    size_t off = 0;
    auto alloc = [&](size_t bytes) -> void* {
        off = (off + 255) & ~(size_t)255;
        void* p = ws + off;
        off += bytes;
        return p;
    };
    const size_t SD = (size_t)4096 * 320;

    float* gnstats = (float*)alloc(64 * sizeof(float));
    h16*   G       = (h16*)alloc(SD * 2);
    h16*   Hf16    = (h16*)alloc(SD * 2);
    float* H       = (float*)alloc(SD * 4);
    h16*   Nh      = (h16*)alloc(SD * 2);        // LN outputs (reused n1/n2/n3)
    h16*   Qb      = (h16*)alloc(SD * 2);        // q1 then q2
    h16*   Kb      = (h16*)alloc(SD * 2);        // k1 then k2
    h16*   Vb      = (h16*)alloc(SD * 2);        // v1 then v2
    h16*   Ab      = (h16*)alloc(SD * 2);        // attn1 then attn2 output
    h16*   Qp      = (h16*)alloc((size_t)8 * 4096 * 64 * 2);
    h16*   Kp      = (h16*)alloc((size_t)8 * 4096 * 64 * 2);
    h16*   Vt      = (h16*)alloc((size_t)8 * 4096 * 64 * 2);
    h16*   K2p     = (h16*)alloc((size_t)8 * 96 * 64 * 2);
    h16*   V2t     = (h16*)alloc((size_t)8 * 64 * 96 * 2);
    float* HG      = (float*)alloc((size_t)4096 * 2560 * 4);
    h16*   FG      = (h16*)alloc((size_t)4096 * 1280 * 2);
    float* P       = (float*)alloc(SD * 4);
    // f16 transposed weight copies (BT layout [N][K])
    h16* pi_w_t = (h16*)alloc((size_t)102400 * 2);
    h16* a1_q_t = (h16*)alloc((size_t)102400 * 2);
    h16* a1_k_t = (h16*)alloc((size_t)102400 * 2);
    h16* a1_v_t = (h16*)alloc((size_t)102400 * 2);
    h16* a1_o_t = (h16*)alloc((size_t)102400 * 2);
    h16* a2_q_t = (h16*)alloc((size_t)102400 * 2);
    h16* a2_k_t = (h16*)alloc((size_t)245760 * 2);
    h16* a2_v_t = (h16*)alloc((size_t)245760 * 2);
    h16* a2_o_t = (h16*)alloc((size_t)102400 * 2);
    h16* ffw1_t = (h16*)alloc((size_t)819200 * 2);
    h16* ffw2_t = (h16*)alloc((size_t)409600 * 2);
    h16* po_w_t = (h16*)alloc((size_t)102400 * 2);
    h16* ctx_h  = (h16*)alloc((size_t)59136 * 2);   // A operand, row-major

    auto cvtT = [&](const float* w, h16* wT, int K, int N) {
        sd_cvt_t_kernel<<<(K * N + 255) / 256, 256, 0, stream>>>(w, wT, K, N);
    };
    cvtT(pi_w, pi_w_t, 320, 320);
    cvtT(a1_q, a1_q_t, 320, 320);  cvtT(a1_k, a1_k_t, 320, 320);
    cvtT(a1_v, a1_v_t, 320, 320);  cvtT(a1_o, a1_o_t, 320, 320);
    cvtT(a2_q, a2_q_t, 320, 320);  cvtT(a2_k, a2_k_t, 768, 320);
    cvtT(a2_v, a2_v_t, 768, 320);  cvtT(a2_o, a2_o_t, 320, 320);
    cvtT(ffw1, ffw1_t, 320, 2560); cvtT(ffw2, ffw2_t, 1280, 320);
    cvtT(po_w, po_w_t, 320, 320);
    sd_f32_to_f16_kernel<<<(59136 + 255) / 256, 256, 0, stream>>>(CTX, ctx_h, 59136);

    auto gemm = [&](const h16* A, const h16* BT, const float* bias, const float* resid,
                    float* outF, h16* outH, int M, int N, int K) {
        dim3 g((N + 63) / 64, (M + 127) / 128);
        sd_gemm_kernel<<<g, 256, 0, stream>>>(A, BT, bias, resid, outF, outH, M, N, K);
    };

    // GroupNorm -> G f16 [4096,320]
    sd_gn_stats_kernel<<<32, 256, 0, stream>>>(X, gnstats);
    sd_gn_apply_kernel<<<(1310720 + 255) / 256, 256, 0, stream>>>(X, gnstats, gn_g, gn_b, G);

    // proj_in: H = G @ pi_w + pi_b
    gemm(G, pi_w_t, pi_b, nullptr, H, nullptr, 4096, 320, 320);

    // ---- self-attention ----
    sd_ln_kernel<<<4096, 32, 0, stream>>>(H, n1_g, n1_b, Nh);
    gemm(Nh, a1_q_t, nullptr, nullptr, nullptr, Qb, 4096, 320, 320);
    gemm(Nh, a1_k_t, nullptr, nullptr, nullptr, Kb, 4096, 320, 320);
    gemm(Nh, a1_v_t, nullptr, nullptr, nullptr, Vb, 4096, 320, 320);
    sd_repack_qkv1_kernel<<<(2097152 + 255) / 256, 256, 0, stream>>>(Qb, Kb, Vb, Qp, Kp, Vt);
    sd_attn1_kernel<<<dim3(64, 8), 128, 0, stream>>>(Qp, Kp, Vt, Ab);
    gemm(Ab, a1_o_t, a1_ob, H, H, nullptr, 4096, 320, 320);   // H += attn1_out

    // ---- cross-attention ----
    sd_ln_kernel<<<4096, 32, 0, stream>>>(H, n2_g, n2_b, Nh);
    gemm(Nh, a2_q_t, nullptr, nullptr, nullptr, Qb, 4096, 320, 320);
    gemm(ctx_h, a2_k_t, nullptr, nullptr, nullptr, Kb, 77, 320, 768);
    gemm(ctx_h, a2_v_t, nullptr, nullptr, nullptr, Vb, 77, 320, 768);
    sd_repack_q_kernel<<<(2097152 + 255) / 256, 256, 0, stream>>>(Qb, Qp);
    sd_repack_kv2_kernel<<<(49152 + 255) / 256, 256, 0, stream>>>(Kb, Vb, K2p, V2t);
    sd_attn2_kernel<<<dim3(64, 8), 128, 0, stream>>>(Qp, K2p, V2t, Ab);
    gemm(Ab, a2_o_t, a2_ob, H, H, nullptr, 4096, 320, 320);   // H += attn2_out

    // ---- GEGLU feed-forward ----
    sd_ln_kernel<<<4096, 32, 0, stream>>>(H, n3_g, n3_b, Nh);
    gemm(Nh, ffw1_t, ffb1, nullptr, HG, nullptr, 4096, 2560, 320);
    sd_geglu_kernel<<<((4096 * 1280) + 255) / 256, 256, 0, stream>>>(HG, FG);
    gemm(FG, ffw2_t, ffb2, H, H, Hf16, 4096, 320, 1280);      // H += ff, also f16 copy

    // ---- proj_out + residual (with transpose back to CHW) ----
    gemm(Hf16, po_w_t, po_b, nullptr, P, nullptr, 4096, 320, 320);
    sd_final_kernel<<<(1310720 + 255) / 256, 256, 0, stream>>>(P, X, OUT);
}
